// EdgeCNN_22857815949374
// MI455X (gfx1250) — matrix operand order repaired and checked by reading, not compiled, not run on hardware
//
#include <hip/hip_runtime.h>
#include <hip/hip_bf16.h>

#define NNODES 20000
#define NF 64
#define KNN 6
#define NH 128

typedef __attribute__((ext_vector_type(16))) _Float16 v16h;
typedef __attribute__((ext_vector_type(8)))  _Float16 v8h;
typedef __attribute__((ext_vector_type(8)))  float    v8f;

__device__ inline v8f wmma16(v16h a, v16h b, v8f c) {
    // D = A(16x32 f16) * B(32x16 f16) + C(16x16 f32)
    return __builtin_amdgcn_wmma_f32_16x16x32_f16(
        /*neg_a=*/false, a, /*neg_b=*/false, b,
        /*c_mod=*/(short)0, c, /*reuse_a=*/false, /*reuse_b=*/false);
}

__device__ inline void wait_ds() { asm volatile("s_wait_dscnt 0" ::: "memory"); }

// Branchless sorted-insert into ascending top-K list (cndmask chain, no exec churn).
__device__ inline void topk_ins(float (&d)[KNN], int (&ix)[KNN], float v, int j) {
#pragma unroll
    for (int s = 0; s < KNN; ++s) {
        bool lt = v < d[s];
        float nd = lt ? v : d[s];
        int   ni = lt ? j : ix[s];
        float nv = lt ? d[s] : v;
        int   nj = lt ? ix[s] : j;
        d[s] = nd; ix[s] = ni; v = nv; j = nj;
    }
}

// A-tile (16 rows x 32 K) from row-major f16 matrix.
// ISA layout: lanes 0-15 hold K=[0..7],[16..23] of row M=lane; lanes 16-31 hold K=[8..15],[24..31].
__device__ inline v16h load_a(const _Float16* base, int stride, int row0, int kb, int lane) {
    const _Float16* p = base + (row0 + (lane & 15)) * stride + kb + ((lane >> 4) << 3);
    v8h lo = *(const v8h*)(p);        // elements 0..7
    v8h hi = *(const v8h*)(p + 16);   // elements 8..15
    v16h a;
#pragma unroll
    for (int e = 0; e < 8; ++e) { a[e] = lo[e]; a[e + 8] = hi[e]; }
    return a;
}

// B-tile (32 K x 16 cols) from column-transposed storage baseT[n][k]:
// lane owns one column; lanes 0-15 hold K=[0..7],[16..23], lanes 16-31 hold K=[8..15],[24..31].
__device__ inline v16h load_b(const _Float16* baseT, int stride, int col0, int kb, int lane) {
    const _Float16* p = baseT + (col0 + (lane & 15)) * stride + kb + ((lane >> 4) << 3);
    v8h lo = *(const v8h*)(p);
    v8h hi = *(const v8h*)(p + 16);
    v16h b;
#pragma unroll
    for (int e = 0; e < 8; ++e) { b[e] = lo[e]; b[e + 8] = hi[e]; }
    return b;
}

// ---------------- prep: f16 copies, row norms, transposed f16 weights ----------------
__global__ __launch_bounds__(256) void prep_kernel(const float* __restrict__ x,
        const float* __restrict__ W1, const float* __restrict__ W2,
        _Float16* __restrict__ xh, float* __restrict__ sq,
        _Float16* __restrict__ w1t, _Float16* __restrict__ w2t) {
    int i = blockIdx.x * blockDim.x + threadIdx.x;
    if (i < NNODES) {
        float s = 0.f;
#pragma unroll 8
        for (int f = 0; f < NF; ++f) {
            float v = x[i * NF + f];
            s += v * v;
            xh[i * NF + f] = (_Float16)v;
        }
        sq[i] = s;
    }
    int total = gridDim.x * blockDim.x;
    for (int e = i; e < NH * NH; e += total) {
        int n = e >> 7, k = e & 127;
        w1t[n * NH + k] = (_Float16)W1[k * NH + n];   // W1 is [2F][H] -> store [H][2F]
        w2t[n * NH + k] = (_Float16)W2[k * NH + n];   // W2 is [H][H]  -> store [H][H]^T
    }
}

// ---------------- kNN: Gram via WMMA (pipelined), per-row top-6 ----------------
#define SDSTR 33   // padded LDS row stride (floats): conflict-free row scans

__global__ __launch_bounds__(32) void knn_wmma_kernel(const _Float16* __restrict__ xh,
        const float* __restrict__ sq, int* __restrict__ knn_idx) {
    __shared__ __align__(16) float s_d[16 * SDSTR];
    __shared__ float s_md[32 * KNN];
    __shared__ int   s_mi[32 * KNN];
    const int lane = threadIdx.x;
    const int mr = blockIdx.x * 16;          // 1250 blocks, exact

    // center-row A tiles held in registers for the whole column sweep
    v16h a0 = load_a(xh, NF, mr, 0,  lane);
    v16h a1 = load_a(xh, NF, mr, 32, lane);

    const int mloc = (lane >> 4) << 3;       // C-layout: M = r + (lane>=16 ? 8 : 0)
    const int cl   = lane & 15;
    float sqi[8];
#pragma unroll
    for (int r = 0; r < 8; ++r) sqi[r] = sq[mr + mloc + r];

    float bd[KNN]; int bi[KNN];
#pragma unroll
    for (int s = 0; s < KNN; ++s) { bd[s] = 3.4e38f; bi[s] = 0; }

    const int myRow = cl;
    const int hw    = lane >> 4;             // half-wave id: scans 16 of the 32 columns

    // software pipeline: B tiles for the NEXT 32 columns load while current compute/scan runs
    v16h b00 = load_b(xh, NF, 0,  0,  lane);
    v16h b01 = load_b(xh, NF, 0,  32, lane);
    v16h b10 = load_b(xh, NF, 16, 0,  lane);
    v16h b11 = load_b(xh, NF, 16, 32, lane);

    for (int cb = 0; cb < NNODES; cb += 32) {      // 625 iterations, exact
        int nb = (cb + 32 < NNODES) ? cb + 32 : 0; // clamped (values unused on last iter)
        v16h n00 = load_b(xh, NF, nb,      0,  lane);
        v16h n01 = load_b(xh, NF, nb,      32, lane);
        v16h n10 = load_b(xh, NF, nb + 16, 0,  lane);
        v16h n11 = load_b(xh, NF, nb + 16, 32, lane);
        float sqj0 = sq[cb + cl];
        float sqj1 = sq[cb + 16 + cl];

        v8f c0 = {}, c1 = {};
        c0 = wmma16(a0, b00, c0);
        c1 = wmma16(a0, b10, c1);   // pair sharing A operand back-to-back
        c0 = wmma16(a1, b01, c0);
        c1 = wmma16(a1, b11, c1);

#pragma unroll
        for (int r = 0; r < 8; ++r) {
            s_d[(mloc + r) * SDSTR + cl]      = sqi[r] + sqj0 - 2.0f * c0[r];
            s_d[(mloc + r) * SDSTR + 16 + cl] = sqi[r] + sqj1 - 2.0f * c1[r];
        }
        wait_ds();
#pragma unroll
        for (int cc = 0; cc < 16; ++cc) {
            float v = s_d[myRow * SDSTR + hw * 16 + cc];
            if (v < bd[KNN - 1])                   // rare divergent guard
                topk_ins(bd, bi, v, cb + hw * 16 + cc);
        }
        b00 = n00; b01 = n01; b10 = n10; b11 = n11;
        // DS ops are in-order per wave: next iteration's stores cannot pass these loads
    }

    // merge the two half-row top-6 lists through LDS
#pragma unroll
    for (int s = 0; s < KNN; ++s) { s_md[lane * KNN + s] = bd[s]; s_mi[lane * KNN + s] = bi[s]; }
    wait_ds();
    if (lane < 16) {
#pragma unroll
        for (int s = 0; s < KNN; ++s) {
            float v = s_md[(lane + 16) * KNN + s];
            int   j = s_mi[(lane + 16) * KNN + s];
            if (v < bd[KNN - 1]) topk_ins(bd, bi, v, j);
        }
#pragma unroll
        for (int s = 0; s < KNN; ++s) knn_idx[(mr + lane) * KNN + s] = bi[s];
    }
}

// ---------------- Edge MLP: msg -> W1/ReLU -> W2 -> max over K ----------------
__global__ __launch_bounds__(128) void mlp_wmma_kernel(const _Float16* __restrict__ xh,
        const int* __restrict__ knn_idx,
        const _Float16* __restrict__ w1t, const float* __restrict__ b1,
        const _Float16* __restrict__ w2t, const float* __restrict__ b2,
        float* __restrict__ out) {
    __shared__ __align__(32) _Float16 s_h1[4][16 * NH];  // layer-1 activations (f16)
    __shared__ __align__(16) float    s_h2[4][16 * NH];  // layer-2 outputs (f32)
    __shared__ float s_mx[4][8 * NH];                    // per-node running max
    const int lane = threadIdx.x & 31;
    const int wave = threadIdx.x >> 5;
    const int nodeBase = (blockIdx.x * 4 + wave) * 8;    // 625 blocks * 4 waves * 8 = 20000 exact
    _Float16* h1 = s_h1[wave];
    float*    h2 = s_h2[wave];
    float*    mx = s_mx[wave];

    for (int i = lane; i < 8 * NH; i += 32) mx[i] = -3.4e38f;

    const int kh8 = (lane >> 4) << 3;   // A-layout half-wave K offset
    const int cl  = lane & 15;

    for (int t = 0; t < 3; ++t) {       // 3 tiles of 16 rows = 48 = 8 nodes * 6 nbrs, exact
        int r    = t * 16 + cl;
        int node = nodeBase + r / KNN;
        int nbr  = knn_idx[node * KNN + (r % KNN)];
        const _Float16* xi = xh + node * NF;
        const _Float16* xj = xh + nbr  * NF;

        // msg A tiles: K chunks 0,1 = x_i ; chunks 2,3 = x_j - x_i (packed f16 subtract)
        v16h A[4];
#pragma unroll
        for (int c = 0; c < 2; ++c) {
            const _Float16* p = xi + c * 32 + kh8;
            v8h lo = *(const v8h*)(p);
            v8h hi = *(const v8h*)(p + 16);
#pragma unroll
            for (int e = 0; e < 8; ++e) { A[c][e] = lo[e]; A[c][e + 8] = hi[e]; }
        }
#pragma unroll
        for (int c = 2; c < 4; ++c) {
            int off = (c - 2) * 32 + kh8;
            v8h lo = *(const v8h*)(xj + off)      - *(const v8h*)(xi + off);
            v8h hi = *(const v8h*)(xj + off + 16) - *(const v8h*)(xi + off + 16);
#pragma unroll
            for (int e = 0; e < 8; ++e) { A[c][e] = lo[e]; A[c][e + 8] = hi[e]; }
        }

        // layer 1: 8 N-tiles x 4 K-steps = 32 WMMA; bias+ReLU into LDS as f16 (A-layout source)
        for (int nt = 0; nt < 8; ++nt) {
            v8f acc = {};
#pragma unroll
            for (int kc = 0; kc < 4; ++kc) {
                v16h b = load_b(w1t, 2 * NF, nt * 16, kc * 32, lane);
                acc = wmma16(A[kc], b, acc);
            }
            float bias = b1[nt * 16 + cl];
#pragma unroll
            for (int rr = 0; rr < 8; ++rr) {
                float v = acc[rr] + bias;
                h1[(kh8 + rr) * NH + nt * 16 + cl] = (_Float16)(v > 0.f ? v : 0.f);
            }
        }
        wait_ds();

        // layer 2: re-swizzle h1 from LDS into A-layout, 32 WMMA
        v16h A2[4];
#pragma unroll
        for (int kc = 0; kc < 4; ++kc) A2[kc] = load_a(h1, NH, 0, kc * 32, lane);
        for (int nt = 0; nt < 8; ++nt) {
            v8f acc = {};
#pragma unroll
            for (int kc = 0; kc < 4; ++kc) {
                v16h b = load_b(w2t, NH, nt * 16, kc * 32, lane);
                acc = wmma16(A2[kc], b, acc);
            }
            float bias = b2[nt * 16 + cl];
#pragma unroll
            for (int rr = 0; rr < 8; ++rr)
                h2[(kh8 + rr) * NH + nt * 16 + cl] = acc[rr] + bias;
        }
        wait_ds();

        // max over K: each lane owns 4 columns -> race-free LDS RMW
#pragma unroll
        for (int cc = 0; cc < 4; ++cc) {
            int col = lane * 4 + cc;
#pragma unroll
            for (int row = 0; row < 16; ++row) {
                int nl = (t * 16 + row) / KNN;           // local node 0..7
                float v = h2[row * NH + col];
                float m = mx[nl * NH + col];
                mx[nl * NH + col] = v > m ? v : m;
            }
        }
        wait_ds();
    }

    for (int i = lane; i < 8 * NH; i += 32)
        out[nodeBase * NH + i] = mx[i];
}

extern "C" void kernel_launch(void* const* d_in, const int* in_sizes, int n_in,
                              void* d_out, int out_size, void* d_ws, size_t ws_size,
                              hipStream_t stream) {
    (void)in_sizes; (void)n_in; (void)out_size; (void)ws_size;
    const float* x  = (const float*)d_in[0];
    // d_in[1] = edge_index (unused by reference)
    const float* W1 = (const float*)d_in[2];
    const float* b1 = (const float*)d_in[3];
    const float* W2 = (const float*)d_in[4];
    const float* b2 = (const float*)d_in[5];
    float* out = (float*)d_out;

    char* ws = (char*)d_ws;                       // ~3.2 MB total
    _Float16* xh  = (_Float16*)(ws);              // N*64 f16   = 2,560,000 B
    float*    sq  = (float*)   (ws + 2560000);    // N f32      =    80,000 B
    int*      knn = (int*)     (ws + 2640000);    // N*6 int    =   480,000 B
    _Float16* w1t = (_Float16*)(ws + 3120000);    // 128*128 f16=    32,768 B
    _Float16* w2t = (_Float16*)(ws + 3152768);    // 128*128 f16=    32,768 B

    prep_kernel<<<(NNODES + 255) / 256, 256, 0, stream>>>(x, W1, W2, xh, sq, w1t, w2t);
    knn_wmma_kernel<<<NNODES / 16, 32, 0, stream>>>(xh, sq, knn);
    mlp_wmma_kernel<<<NNODES / 32, 128, 0, stream>>>(xh, knn, w1t, b1, w2t, b2, out);
}
